// SAGE_PyG_13039520710798
// MI455X (gfx1250) — matrix operand order repaired and checked by reading, not compiled, not run on hardware
//
#include <hip/hip_runtime.h>

#define F_HID 128
#define BN_EPS 1e-5f

typedef __attribute__((ext_vector_type(16))) __bf16 v16bf;
typedef __attribute__((ext_vector_type(8)))  float v8f;

union FragBF {
    v16bf v;
    unsigned int d[8];
};

// round-to-nearest-even f32 -> bf16, pack pair (elem k in [15:0], k+1 in [31:16])
__device__ __forceinline__ unsigned int pack2_bf16(float a, float b) {
    unsigned int ua = __builtin_bit_cast(unsigned int, a);
    unsigned int ub = __builtin_bit_cast(unsigned int, b);
    ua = (ua + 0x7FFFu + ((ua >> 16) & 1u)) >> 16;
    ub = (ub + 0x7FFFu + ((ub >> 16) & 1u)) >> 16;
    return (ub << 16) | (ua & 0xFFFFu);
}

__device__ __forceinline__ unsigned short to_bf16(float a) {
    unsigned int ua = __builtin_bit_cast(unsigned int, a);
    return (unsigned short)((ua + 0x7FFFu + ((ua >> 16) & 1u)) >> 16);
}

__device__ __forceinline__ float bf16_lo(unsigned int u) {
    return __builtin_bit_cast(float, u << 16);
}
__device__ __forceinline__ float bf16_hi(unsigned int u) {
    return __builtin_bit_cast(float, u & 0xFFFF0000u);
}

// Build a 16xbf16 fragment from two contiguous uint4 loads.
__device__ __forceinline__ v16bf load_frag(const uint4* base, int i0, int i1) {
    FragBF F;
    uint4 lo = base[i0], hi = base[i1];
    F.d[0] = lo.x; F.d[1] = lo.y; F.d[2] = lo.z; F.d[3] = lo.w;
    F.d[4] = hi.x; F.d[5] = hi.y; F.d[6] = hi.z; F.d[7] = hi.w;
    return F.v;
}

__global__ void fill_zero(float* __restrict__ p, long long n) {
    long long i = (long long)blockIdx.x * blockDim.x + threadIdx.x;
    if (i < n) p[i] = 0.0f;
}

__global__ void count_edges(const int* __restrict__ dst, float* __restrict__ cnt, int E) {
    int e = blockIdx.x * blockDim.x + threadIdx.x;
    if (e < E) atomicAdd(&cnt[dst[e]], 1.0f);
}

__global__ void invert_cnt(float* __restrict__ cnt, int N) {
    int i = blockIdx.x * blockDim.x + threadIdx.x;
    if (i < N) cnt[i] = 1.0f / fmaxf(cnt[i], 1.0f);
}

// f32 [n2*2] -> packed bf16 dwords [n2] (streaming, mostly L2-resident)
__global__ void cvt_f32_bf16(const float* __restrict__ in, unsigned int* __restrict__ out,
                             long long n2) {
    long long i = (long long)blockIdx.x * blockDim.x + threadIdx.x;
    if (i < n2) {
        float2 v = ((const float2*)in)[i];
        out[i] = pack2_bf16(v.x, v.y);
    }
}

// Pre-pack weight matrix W [128 x FOUT] (row-major f32) into WMMA B-fragment lane order:
// out[((ct*4 + ks)*32 + lane)*8 + p] = bf16pair( W[kb, ct*16 + n], W[kb+1, ...] )
// with kb = ks*32 + (p>>2)*16 + (lane>>4)*8 + (p&3)*2, n = lane&15 (B 32x16 layout per ISA 7.12.5 family).
__global__ void pack_weights(const float* __restrict__ W, unsigned int* __restrict__ out,
                             int FOUT) {
    int t = blockIdx.x * blockDim.x + threadIdx.x;
    int total = (FOUT / 16) * 4 * 32 * 8;
    if (t >= total) return;
    int p    = t & 7;
    int lane = (t >> 3) & 31;
    int ks   = (t >> 8) & 3;
    int ct   = t >> 10;
    int half = lane >> 4;
    int n    = lane & 15;
    int kb   = ks * 32 + (p >> 2) * 16 + half * 8 + (p & 3) * 2;
    int col  = ct * 16 + n;
    out[t] = pack2_bf16(W[(size_t)kb * FOUT + col], W[(size_t)(kb + 1) * FOUT + col]);
}

// One wave per edge. Features stored as packed bf16 (64 dwords / row): each lane
// gathers 2 dwords (4 feats, 256B/row per wave, coalesced), widens to f32, and does
// 4x global_atomic_add_f32 into the L2-resident f32 accumulator.
__global__ void scatter_add(const unsigned int* __restrict__ h, const int* __restrict__ src,
                            const int* __restrict__ dst, float* __restrict__ agg, int E) {
    int t = blockIdx.x * blockDim.x + threadIdx.x;
    int e = t >> 5;
    int lane = t & 31;
    if (e >= E) return;
    int s = src[e];
    int d = dst[e];
    const uint2* row = (const uint2*)(h + (size_t)s * (F_HID / 2));
    uint2 v = row[lane];
    if (e + 4 < E) {
        // warm an upcoming gather row (global_prefetch_b8)
        __builtin_prefetch(h + (size_t)src[e + 4] * (F_HID / 2) + lane * 2, 0, 0);
    }
    float* arow = agg + (size_t)d * F_HID + lane * 4;
    atomicAdd(arow + 0, bf16_lo(v.x));
    atomicAdd(arow + 1, bf16_hi(v.x));
    atomicAdd(arow + 2, bf16_lo(v.y));
    atomicAdd(arow + 3, bf16_hi(v.y));
}

// Fused SAGE layer: out = inv_row * (aggb @ Wl) + h @ Wr + bias [+ BN + ReLU]
// (mean division algebraically hoisted to the epilogue since inv is a row scalar).
// Block: 16 node rows; one wave per 16-col tile; K=256 total -> 8x v_wmma_f32_16x16x32_bf16.
// All A/B fragments are native packed bf16 -> pure b128 loads, zero conversion VALU.
template <int FOUT, bool BN, bool OUT_BF16>
__global__ __launch_bounds__(32 * (FOUT / 16)) void sage_gemm(
    const unsigned int* __restrict__ hb, const unsigned int* __restrict__ aggb,
    const float* __restrict__ inv, const unsigned int* __restrict__ Wlp,
    const unsigned int* __restrict__ Wrp, const float* __restrict__ bias,
    const float* __restrict__ g, const float* __restrict__ be,
    const float* __restrict__ rm, const float* __restrict__ rv,
    void* __restrict__ hout, int N) {
    const int lane = threadIdx.x & 31;
    const int wave = threadIdx.x >> 5;
    const int m    = lane & 15;
    const int half = lane >> 4;
    const int row0 = blockIdx.x * 16;
    const int col0 = wave * 16;

    int arow = row0 + m;
    if (arow > N - 1) arow = N - 1;          // clamp (keeps EXEC all-1s for WMMA)
    const uint4* arow4 = (const uint4*)(aggb + (size_t)arow * (F_HID / 2)); // 16 uint4 / row
    const uint4* hrow4 = (const uint4*)(hb + (size_t)arow * (F_HID / 2));
    const uint4* wl4   = (const uint4*)Wlp;
    const uint4* wr4   = (const uint4*)Wrp;

    v8f c1 = {}, c2 = {};

    // Part 1: aggb @ Wl (K = 0..127) -> c1 (scaled by inv in epilogue)
#pragma unroll
    for (int ks = 0; ks < 4; ++ks) {
        v16bf A = load_frag(arow4, ks * 4 + half, ks * 4 + 2 + half);
        int bi = ((wave * 4 + ks) * 32 + lane) * 2;
        v16bf B = load_frag(wl4, bi, bi + 1);
        c1 = __builtin_amdgcn_wmma_f32_16x16x32_bf16(false, A, false, B,
                                                     (short)0, c1, false, false);
    }
    // Part 2: h @ Wr (K = 0..127) -> c2
#pragma unroll
    for (int ks = 0; ks < 4; ++ks) {
        v16bf A = load_frag(hrow4, ks * 4 + half, ks * 4 + 2 + half);
        int bi = ((wave * 4 + ks) * 32 + lane) * 2;
        v16bf B = load_frag(wr4, bi, bi + 1);
        c2 = __builtin_amdgcn_wmma_f32_16x16x32_bf16(false, A, false, B,
                                                     (short)0, c2, false, false);
    }

    // Epilogue. C/D layout: VGPR r holds M = r (lanes 0-15) / M = r+8 (lanes 16-31); N = lane&15.
    const int col = col0 + m;
    float bb = bias[col];
    float scale = 1.0f, shift = 0.0f;
    if constexpr (BN) {
        scale = g[col] * rsqrtf(rv[col] + BN_EPS);
        shift = be[col] - rm[col] * scale;
    }
#pragma unroll
    for (int r = 0; r < 8; ++r) {
        int row = row0 + r + half * 8;
        if (row < N) {
            float v = fmaf(inv[row], c1[r], c2[r]) + bb;
            if constexpr (BN) v = fmaxf(v * scale + shift, 0.0f);
            if constexpr (OUT_BF16) {
                ((unsigned short*)hout)[(size_t)row * FOUT + col] = to_bf16(v);
            } else {
                ((float*)hout)[(size_t)row * FOUT + col] = v;
            }
        }
    }
}

extern "C" void kernel_launch(void* const* d_in, const int* in_sizes, int n_in,
                              void* d_out, int out_size, void* d_ws, size_t ws_size,
                              hipStream_t stream) {
    (void)n_in; (void)out_size; (void)ws_size;

    const float* x   = (const float*)d_in[0];
    const int*   ei  = (const int*)d_in[1];
    const float* W0l = (const float*)d_in[2];
    const float* b0  = (const float*)d_in[3];
    const float* W0r = (const float*)d_in[4];
    const float* g0  = (const float*)d_in[5];
    const float* be0 = (const float*)d_in[6];
    const float* rm0 = (const float*)d_in[7];
    const float* rv0 = (const float*)d_in[8];
    const float* W1l = (const float*)d_in[9];
    const float* b1  = (const float*)d_in[10];
    const float* W1r = (const float*)d_in[11];
    const float* g1  = (const float*)d_in[12];
    const float* be1 = (const float*)d_in[13];
    const float* rm1 = (const float*)d_in[14];
    const float* rv1 = (const float*)d_in[15];
    const float* W2l = (const float*)d_in[16];
    const float* b2  = (const float*)d_in[17];
    const float* W2r = (const float*)d_in[18];

    const int N = in_sizes[0] / F_HID;
    const int E = in_sizes[1] / 2;
    const int* src = ei;
    const int* dst = ei + E;

    // workspace layout (all 16B-aligned):
    // inv[Npad] f32 | agg[N*128] f32 | aggb[N*64] u32 | xb[N*64] u32 | hA[N*64] u32 |
    // hB[N*64] u32 | packed weights (~160KB)
    float* inv = (float*)d_ws;
    size_t npad = ((size_t)N + 63) & ~(size_t)63;
    float* agg = inv + npad;
    unsigned int* aggb = (unsigned int*)(agg + (size_t)N * F_HID);
    unsigned int* xb = aggb + (size_t)N * (F_HID / 2);
    unsigned int* hA = xb + (size_t)N * (F_HID / 2);
    unsigned int* hB = hA + (size_t)N * (F_HID / 2);
    unsigned int* wp = hB + (size_t)N * (F_HID / 2);
    const int WP128 = 8 * 4 * 32 * 8;   // 8192 dwords for a 128x128 matrix
    const int WP64  = 4 * 4 * 32 * 8;   // 4096 dwords for a 128x64 matrix
    unsigned int* W0lp = wp;            unsigned int* W0rp = W0lp + WP128;
    unsigned int* W1lp = W0rp + WP128;  unsigned int* W1rp = W1lp + WP128;
    unsigned int* W2lp = W1rp + WP128;  unsigned int* W2rp = W2lp + WP64;

    const int TB = 256;
    long long nf = (long long)N * F_HID;
    dim3 gridN((unsigned)((N + TB - 1) / TB));
    dim3 gridNF((unsigned)((nf + TB - 1) / TB));
    dim3 gridN2((unsigned)((nf / 2 + TB - 1) / TB));
    dim3 gridE((unsigned)((E + TB - 1) / TB));
    dim3 gridScat((unsigned)(((long long)E * 32 + TB - 1) / TB));
    dim3 gridGemm((unsigned)((N + 15) / 16));
    dim3 gridWP128((WP128 + TB - 1) / TB), gridWP64((WP64 + TB - 1) / TB);

    // One-time prep: degree inverse, x -> bf16, weights -> fragment-packed bf16.
    fill_zero<<<gridN, TB, 0, stream>>>(inv, N);
    count_edges<<<gridE, TB, 0, stream>>>(dst, inv, E);
    invert_cnt<<<gridN, TB, 0, stream>>>(inv, N);
    cvt_f32_bf16<<<gridN2, TB, 0, stream>>>(x, xb, nf / 2);
    pack_weights<<<gridWP128, TB, 0, stream>>>(W0l, W0lp, 128);
    pack_weights<<<gridWP128, TB, 0, stream>>>(W0r, W0rp, 128);
    pack_weights<<<gridWP128, TB, 0, stream>>>(W1l, W1lp, 128);
    pack_weights<<<gridWP128, TB, 0, stream>>>(W1r, W1rp, 128);
    pack_weights<<<gridWP64, TB, 0, stream>>>(W2l, W2lp, 64);
    pack_weights<<<gridWP64, TB, 0, stream>>>(W2r, W2rp, 64);

    // Layer 0: xb -> hA (BN + ReLU, bf16 out)
    fill_zero<<<gridNF, TB, 0, stream>>>(agg, nf);
    scatter_add<<<gridScat, TB, 0, stream>>>(xb, src, dst, agg, E);
    cvt_f32_bf16<<<gridN2, TB, 0, stream>>>(agg, aggb, nf / 2);
    sage_gemm<128, true, true><<<gridGemm, 256, 0, stream>>>(
        xb, aggb, inv, W0lp, W0rp, b0, g0, be0, rm0, rv0, hA, N);

    // Layer 1: hA -> hB (BN + ReLU, bf16 out)
    fill_zero<<<gridNF, TB, 0, stream>>>(agg, nf);
    scatter_add<<<gridScat, TB, 0, stream>>>(hA, src, dst, agg, E);
    cvt_f32_bf16<<<gridN2, TB, 0, stream>>>(agg, aggb, nf / 2);
    sage_gemm<128, true, true><<<gridGemm, 256, 0, stream>>>(
        hA, aggb, inv, W1lp, W1rp, b1, g1, be1, rm1, rv1, hB, N);

    // Layer 2: hB -> d_out (64 cols, no BN/ReLU, f32 out)
    fill_zero<<<gridNF, TB, 0, stream>>>(agg, nf);
    scatter_add<<<gridScat, TB, 0, stream>>>(hB, src, dst, agg, E);
    cvt_f32_bf16<<<gridN2, TB, 0, stream>>>(agg, aggb, nf / 2);
    sage_gemm<64, false, false><<<gridGemm, 128, 0, stream>>>(
        hB, aggb, inv, W2lp, W2rp, b2, nullptr, nullptr, nullptr, nullptr, d_out, N);
}
